// Seq2SeqTransformer_85796266705289
// MI455X (gfx1250) — compile-verified, matrix operand order checked by reading, not run on hardware
//
#include <hip/hip_runtime.h>
#include <hip/hip_bf16.h>
#include <math.h>

// ---------------- model dims (fixed by reference) ----------------
#define L_LAYERS 6
#define D_MODEL  1024
#define FF_DIM   4096
#define SEQ_LEN  512
#define BATCH    8
#define VOCAB    32004

typedef __attribute__((ext_vector_type(16))) __bf16 v16bf;
typedef __attribute__((ext_vector_type(8)))  __bf16 v8bf;
typedef __attribute__((ext_vector_type(8)))  float  v8f;
typedef __attribute__((ext_vector_type(4)))  int    v4i;

union BF16x16 { v16bf v; v8bf h[2]; };

static __device__ __forceinline__ unsigned short f32_to_bf16_rne(float f) {
    union { float f; unsigned int u; } c; c.f = f;
    unsigned int u = c.u;
    u += 0x7FFFu + ((u >> 16) & 1u);   // round to nearest even
    return (unsigned short)(u >> 16);
}

// 16-byte global->LDS copy; async DMA on gfx1250 (tracked by ASYNCcnt),
// falls back to a plain load+store pair if the builtin is absent.
// Builtin prototype (from hipcc diagnostic): param1 is v4i* (global src);
// LDS destination passed as address_space(3) v4i* (explicit cast emits
// addrspacecast; converts implicitly if the parameter is generic).
static __device__ __forceinline__
void async_cp16(unsigned short* lds_dst, const unsigned short* g_src) {
#if __has_builtin(__builtin_amdgcn_global_load_async_to_lds_b128)
    __builtin_amdgcn_global_load_async_to_lds_b128(
        (v4i*)(g_src),
        (__attribute__((address_space(3))) v4i*)(lds_dst),
        0, 0);
#else
    *(int4*)lds_dst = *(const int4*)g_src;
#endif
}

// =====================================================================
// Batched GEMM:  C[b] = relu?( alpha * A[b] x Bt[b]^T + bias )
//   A : M x K row-major bf16 (lda = K)
//   Bt: N x K row-major bf16 (every B operand is pre-transposed so the
//       K-axis is contiguous -> 16B fragment chunks are contiguous in HBM)
//   C : M x N row-major fp32
// Block: 128 threads = 4 waves; block tile 128x128; wave tile 64x64
//  -> 16 x v_wmma_f32_16x16x32_bf16 per wave per 32-wide K step.
// LDS tiles are stored directly in the 16-bit WMMA fragment layout
// (CDNA5 ISA 7.12.2): fragment element h maps to k as
//   h<8 : k = (lane>=16 ? 8 : 0) + h          (16B contiguous chunk)
//   h>=8: k = 16 + (lane>=16 ? 8 : 0) + (h-8) (16B contiguous chunk)
// so each lane slot is filled by two async b128 copies straight from HBM.
// Double-buffered: tile it+1 is issued before waiting ASYNCcnt<=8 for
// tile it (async loads retire in order), overlapping DMA with WMMA.
// =====================================================================
#define BM 128
#define BN 128
#define BK 32
#define LANE_H 40                    // halves per lane slot (80 B, 16B-aligned)
#define TILE_H (8 * 32 * LANE_H)     // halves per LDS tile buffer

__global__ __launch_bounds__(128)
void gemm_bf16t_wmma(const unsigned short* __restrict__ Ag,
                     const unsigned short* __restrict__ Btg,
                     const float* __restrict__ bias, float* __restrict__ Cg,
                     int M, int N, int K,
                     long long strideA, long long strideB, long long strideC,
                     float alpha, int relu)
{
    __shared__ unsigned short ldsA[2][TILE_H];   // 2 x 20 KB
    __shared__ unsigned short ldsB[2][TILE_H];   // 2 x 20 KB

    const int tid  = threadIdx.x;
    const int lane = tid & 31;
    const int wave = tid >> 5;
    const int wm   = wave >> 1;   // 0..1 : 64-row half of the block tile
    const int wn   = wave & 1;    // 0..1 : 64-col half

    const int bm0 = blockIdx.y * BM;
    const int bn0 = blockIdx.x * BN;

    const unsigned short* A  = Ag  + (long long)blockIdx.z * strideA;
    const unsigned short* Bt = Btg + (long long)blockIdx.z * strideB;
    float*                C  = Cg  + (long long)blockIdx.z * strideC;

    // stage one 128x32 A tile + 32x128 B tile into fragment-ordered LDS.
    // 512 16B chunks per tile; each of 128 threads issues 4 A + 4 B copies.
    auto stage = [&](int kk, int buf) {
        #pragma unroll
        for (int c = 0; c < 4; ++c) {
            const int chunk = tid * 4 + c;          // 0..511
            const int t  = chunk >> 6;              // 16-row/col tile
            const int l  = (chunk >> 1) & 31;       // lane slot
            const int hs = chunk & 1;               // half-select (h 0..7 / 8..15)
            const int ro = t * 16 + (l & 15);       // row within block tile
            const int kg = kk + hs * 16 + ((l & 16) ? 8 : 0);
            const long long goff = (long long)K * ro + kg;
            unsigned short* la = &ldsA[buf][(t * 32 + l) * LANE_H + hs * 8];
            unsigned short* lb = &ldsB[buf][(t * 32 + l) * LANE_H + hs * 8];
            async_cp16(la, A  + (long long)bm0 * K + goff);
            async_cp16(lb, Bt + (long long)bn0 * K + goff);
        }
    };

    v8f acc[4][4] = {};
    const int nIter = K / BK;

    stage(0, 0);
    for (int it = 0; it < nIter; ++it) {
        const int cur = it & 1;
        if (it + 1 < nIter) {
            stage((it + 1) * BK, cur ^ 1);
            // current tile's 8 async copies are the oldest outstanding ones
            asm volatile("s_wait_asynccnt 0x8" ::: "memory");
        } else {
            asm volatile("s_wait_asynccnt 0x0" ::: "memory");
        }
        __syncthreads();   // all waves' tile-it data visible in LDS

        BF16x16 af[4], bf[4];
        #pragma unroll
        for (int i = 0; i < 4; ++i) {
            const unsigned short* pa = &ldsA[cur][((wm * 4 + i) * 32 + lane) * LANE_H];
            af[i].h[0] = *(const v8bf*)(pa);
            af[i].h[1] = *(const v8bf*)(pa + 8);
            const unsigned short* pb = &ldsB[cur][((wn * 4 + i) * 32 + lane) * LANE_H];
            bf[i].h[0] = *(const v8bf*)(pb);
            bf[i].h[1] = *(const v8bf*)(pb + 8);
        }
        #pragma unroll
        for (int i = 0; i < 4; ++i)
            #pragma unroll
            for (int j = 0; j < 4; ++j)
                acc[i][j] = __builtin_amdgcn_wmma_f32_16x16x32_bf16(
                    false, af[i].v, false, bf[j].v, (short)0, acc[i][j],
                    false, false);
        __syncthreads();   // done reading buf 'cur' before it is re-staged
    }

    // epilogue: C/D layout = VGPR r -> row r (+8 for lanes 16-31)
    #pragma unroll
    for (int i = 0; i < 4; ++i) {
        const int mbase = bm0 + wm * 64 + i * 16 + ((lane >= 16) ? 8 : 0);
        #pragma unroll
        for (int j = 0; j < 4; ++j) {
            const int n  = bn0 + wn * 64 + j * 16 + (lane & 15);
            const float bv = bias ? bias[n] : 0.0f;
            #pragma unroll
            for (int r = 0; r < 8; ++r) {
                float v = alpha * acc[i][j][r] + bv;
                if (relu) v = fmaxf(v, 0.0f);
                C[(long long)(mbase + r) * N + n] = v;
            }
        }
    }
}

// ============ fp32 -> bf16 elementwise (n even) ============
__global__ __launch_bounds__(256)
void cvt_f32_bf16(const float* __restrict__ in, unsigned short* __restrict__ out,
                  long long n)
{
    const long long i = ((long long)blockIdx.x * 256 + threadIdx.x) * 2;
    if (i >= n) return;
    const unsigned int p = (unsigned int)f32_to_bf16_rne(in[i]) |
                           ((unsigned int)f32_to_bf16_rne(in[i + 1]) << 16);
    *(unsigned int*)&out[i] = p;
}

// ====== fp32 (R x C row-major) -> bf16 transposed (C x R), batched ======
__global__ __launch_bounds__(256)
void cvt_T_f32_bf16(const float* __restrict__ in, unsigned short* __restrict__ out,
                    int R, int C, long long strideIn, long long strideOut)
{
    __shared__ float tile[32][33];
    const float*    src = in  + (long long)blockIdx.z * strideIn;
    unsigned short* dst = out + (long long)blockIdx.z * strideOut;
    const int c0 = blockIdx.x * 32, r0 = blockIdx.y * 32;
    const int tx = threadIdx.x & 31, ty = threadIdx.x >> 5;   // 32 x 8
    #pragma unroll
    for (int i = 0; i < 32; i += 8)
        tile[ty + i][tx] = src[(long long)(r0 + ty + i) * C + c0 + tx];
    __syncthreads();
    #pragma unroll
    for (int i = 0; i < 32; i += 8)
        dst[(long long)(c0 + ty + i) * R + r0 + tx] = f32_to_bf16_rne(tile[tx][ty + i]);
}

// ================= softmax over rows of length 512 =================
__global__ __launch_bounds__(128)
void softmax512(const float* __restrict__ in, float* __restrict__ out)
{
    __shared__ float red[128];
    const long long row = blockIdx.x;
    const float* src = in  + row * SEQ_LEN;
    float*       dst = out + row * SEQ_LEN;
    const int tid = threadIdx.x;

    float v[4];
    float mx = -1e30f;
    #pragma unroll
    for (int i = 0; i < 4; ++i) { v[i] = src[tid + i * 128]; mx = fmaxf(mx, v[i]); }
    red[tid] = mx; __syncthreads();
    for (int s = 64; s > 0; s >>= 1) {
        if (tid < s) red[tid] = fmaxf(red[tid], red[tid + s]);
        __syncthreads();
    }
    mx = red[0]; __syncthreads();

    float sum = 0.0f;
    #pragma unroll
    for (int i = 0; i < 4; ++i) { v[i] = __expf(v[i] - mx); sum += v[i]; }
    red[tid] = sum; __syncthreads();
    for (int s = 64; s > 0; s >>= 1) {
        if (tid < s) red[tid] += red[tid + s];
        __syncthreads();
    }
    const float inv = 1.0f / red[0];
    #pragma unroll
    for (int i = 0; i < 4; ++i) dst[tid + i * 128] = v[i] * inv;
}

// ========== out = LayerNorm(x + r) * g + b over rows of 1024 ==========
__global__ __launch_bounds__(256)
void add_layernorm(const float* __restrict__ x, const float* __restrict__ r,
                   const float* __restrict__ g, const float* __restrict__ b,
                   float* __restrict__ out)
{
    __shared__ float red[256];
    const long long row = blockIdx.x;
    const float* xs = x + row * D_MODEL;
    const float* rs = r + row * D_MODEL;
    float*       os = out + row * D_MODEL;
    const int tid = threadIdx.x;

    float v[4];
    float s = 0.0f;
    #pragma unroll
    for (int i = 0; i < 4; ++i) { v[i] = xs[tid + i * 256] + rs[tid + i * 256]; s += v[i]; }
    red[tid] = s; __syncthreads();
    for (int st = 128; st > 0; st >>= 1) {
        if (tid < st) red[tid] += red[tid + st];
        __syncthreads();
    }
    const float mu = red[0] * (1.0f / (float)D_MODEL); __syncthreads();

    float vs = 0.0f;
    #pragma unroll
    for (int i = 0; i < 4; ++i) { const float d = v[i] - mu; vs += d * d; }
    red[tid] = vs; __syncthreads();
    for (int st = 128; st > 0; st >>= 1) {
        if (tid < st) red[tid] += red[tid + st];
        __syncthreads();
    }
    const float rstd = rsqrtf(red[0] * (1.0f / (float)D_MODEL) + 1e-5f);

    #pragma unroll
    for (int i = 0; i < 4; ++i) {
        const int c = tid + i * 256;
        os[c] = (v[i] - mu) * rstd * g[c] + b[c];
    }
}

// ============ embedding lookup + sinusoidal positional enc ============
__global__ __launch_bounds__(256)
void embed_pos(const int* __restrict__ idx, const float* __restrict__ emb,
               float* __restrict__ out)
{
    const long long i = (long long)blockIdx.x * blockDim.x + threadIdx.x;
    const long long total = (long long)BATCH * SEQ_LEN * D_MODEL;
    if (i >= total) return;
    const int d        = (int)(i % D_MODEL);
    const long long bn = i / D_MODEL;
    const int n        = (int)(bn % SEQ_LEN);
    int token = idx[bn];
    if (token >= VOCAB || token < 0) token = 1;  // UNK
    const float e = emb[(long long)token * D_MODEL + d];
    const float power = (float)(2 * (d >> 1)) * (1.0f / (float)D_MODEL);
    const float inv_denom = exp2f(-power * 13.287712379549449f);  // 1/10000^power
    const float ang = (float)n * inv_denom;
    const float pe  = (d & 1) ? __cosf(ang) : __sinf(ang);
    out[i] = e + pe;
}

// =====================================================================
extern "C" void kernel_launch(void* const* d_in, const int* in_sizes, int n_in,
                              void* d_out, int out_size, void* d_ws, size_t ws_size,
                              hipStream_t stream)
{
    (void)in_sizes; (void)n_in; (void)out_size; (void)ws_size;

    const int*   in_idx  = (const int*)d_in[0];
    const int*   tgt_idx = (const int*)d_in[1];
    const float* emb     = (const float*)d_in[2];
    const float* E[16];  for (int i = 0; i < 16; ++i) E[i] = (const float*)d_in[3 + i];
    const float* Dp[22]; for (int i = 0; i < 22; ++i) Dp[i] = (const float*)d_in[19 + i];

    enum { EWq, EBq, EWk, EBk, EWv, EBv, EWu, EBu, ELn1g, ELn1b,
           EW1, EB1, EW2, EB2, ELn2g, ELn2b };
    enum { DWq, DBq, DWk, DBk, DWv, DBv, DWu, DBu, DLn1g, DLn1b,
           DWedq, DBedq, DWedu, DBedu, DLn2g, DLn2b,
           DW1, DB1, DW2, DB2, DLn3g, DLn3b };

    const long long WQ  = (long long)D_MODEL * D_MODEL;
    const long long W1S = (long long)D_MODEL * FF_DIM;
    const long long BD = D_MODEL, BF = FF_DIM;

    // ---- workspace layout ----
    float* ws = (float*)d_ws;
    const long long SZF = (long long)BATCH * SEQ_LEN * D_MODEL;  // 4,194,304
    float* X   = ws + 0 * SZF;
    float* Qb  = ws + 1 * SZF;
    float* Kb  = ws + 2 * SZF;
    float* Vb  = ws + 3 * SZF;
    float* N1  = ws + 4 * SZF;
    float* N2  = ws + 5 * SZF;
    float* EK  = ws + 6 * SZF;                 // last-encoder-layer K
    float* EV  = ws + 7 * SZF;                 // last-encoder-layer V
    float* FFb = ws + 8 * SZF;                 // 4*SZF (B*512*4096)
    float* Sb  = ws + 12 * SZF;                // B*512*512 attention probs
    unsigned short* Abf = (unsigned short*)(ws + 12 * SZF + SZF / 2);  // 16M halves
    unsigned short* Bbf = (unsigned short*)(ws + 14 * SZF + SZF / 2);  //  4M halves

    float* OUT_Y = (float*)d_out;          // (B,M,D)
    float* OUT_S = (float*)d_out + SZF;    // (B,M,N)

    const long long sQK = (long long)SEQ_LEN * D_MODEL;
    const long long sSS = (long long)SEQ_LEN * SEQ_LEN;
    const int ROWS = BATCH * SEQ_LEN;                     // 4096
    const long long ACT = SZF, SCO = (long long)BATCH * sSS, FFA = 4 * SZF;

    auto cvt = [&](const float* in, unsigned short* out, long long n) {
        cvt_f32_bf16<<<dim3((unsigned)((n / 2 + 255) / 256)), 256, 0, stream>>>(in, out, n);
    };
    auto cvtT = [&](const float* in, unsigned short* out, int R, int C,
                    int batch, long long sIn, long long sOut) {
        cvt_T_f32_bf16<<<dim3(C / 32, R / 32, batch), 256, 0, stream>>>(in, out, R, C, sIn, sOut);
    };
    auto gemm = [&](const unsigned short* A, const unsigned short* Bt, const float* bias,
                    float* C, int M, int N, int K, float alpha, int relu,
                    int batch, long long sA, long long sB, long long sC) {
        gemm_bf16t_wmma<<<dim3(N / BN, M / BM, batch), 128, 0, stream>>>(
            A, Bt, bias, C, M, N, K, sA, sB, sC, alpha, relu);
    };
    const float AATT = 0.125f;   // 1/sqrt(DH) = 1/8

    // ===================== encoder stack =====================
    embed_pos<<<dim3((unsigned)((SZF + 255) / 256)), 256, 0, stream>>>(in_idx, emb, X);
    for (int i = 0; i < L_LAYERS; ++i) {
        float* kp = (i == L_LAYERS - 1) ? EK : Kb;
        float* vp = (i == L_LAYERS - 1) ? EV : Vb;
        cvt(X, Abf, ACT);
        cvtT(E[EWq] + i * WQ, Bbf, D_MODEL, D_MODEL, 1, 0, 0);
        gemm(Abf, Bbf, E[EBq] + i * BD, Qb, ROWS, D_MODEL, D_MODEL, 1.f, 0, 1, 0, 0, 0);
        cvtT(E[EWk] + i * WQ, Bbf, D_MODEL, D_MODEL, 1, 0, 0);
        gemm(Abf, Bbf, E[EBk] + i * BD, kp, ROWS, D_MODEL, D_MODEL, 1.f, 0, 1, 0, 0, 0);
        cvtT(E[EWv] + i * WQ, Bbf, D_MODEL, D_MODEL, 1, 0, 0);
        gemm(Abf, Bbf, E[EBv] + i * BD, vp, ROWS, D_MODEL, D_MODEL, 1.f, 0, 1, 0, 0, 0);
        // attention: S = softmax(alpha * Q K^T) ; K already N x K layout
        cvt(Qb, Abf, ACT); cvt(kp, Bbf, ACT);
        gemm(Abf, Bbf, nullptr, Sb, SEQ_LEN, SEQ_LEN, D_MODEL, AATT, 0, BATCH, sQK, sQK, sSS);
        softmax512<<<dim3(ROWS), 128, 0, stream>>>(Sb, Sb);
        cvt(Sb, Abf, SCO); cvtT(vp, Bbf, SEQ_LEN, D_MODEL, BATCH, sQK, sQK);
        gemm(Abf, Bbf, nullptr, Qb, SEQ_LEN, D_MODEL, SEQ_LEN, 1.f, 0, BATCH, sSS, sQK, sQK);
        cvt(Qb, Abf, ACT); cvtT(E[EWu] + i * WQ, Bbf, D_MODEL, D_MODEL, 1, 0, 0);
        gemm(Abf, Bbf, E[EBu] + i * BD, Vb, ROWS, D_MODEL, D_MODEL, 1.f, 0, 1, 0, 0, 0);
        add_layernorm<<<dim3(ROWS), 256, 0, stream>>>(X, Vb, E[ELn1g] + i * BD, E[ELn1b] + i * BD, N1);
        // FFN
        cvt(N1, Abf, ACT); cvtT(E[EW1] + i * W1S, Bbf, D_MODEL, FF_DIM, 1, 0, 0);
        gemm(Abf, Bbf, E[EB1] + i * BF, FFb, ROWS, FF_DIM, D_MODEL, 1.f, 1, 1, 0, 0, 0);
        cvt(FFb, Abf, FFA); cvtT(E[EW2] + i * W1S, Bbf, FF_DIM, D_MODEL, 1, 0, 0);
        gemm(Abf, Bbf, E[EB2] + i * BD, Qb, ROWS, D_MODEL, FF_DIM, 1.f, 0, 1, 0, 0, 0);
        add_layernorm<<<dim3(ROWS), 256, 0, stream>>>(N1, Qb, E[ELn2g] + i * BD, E[ELn2b] + i * BD, X);
    }

    // ===================== decoder stack =====================
    embed_pos<<<dim3((unsigned)((SZF + 255) / 256)), 256, 0, stream>>>(tgt_idx, emb, X);
    for (int i = 0; i < L_LAYERS; ++i) {
        const bool last = (i == L_LAYERS - 1);
        // self-attention (reference has no causal mask)
        cvt(X, Abf, ACT);
        cvtT(Dp[DWq] + i * WQ, Bbf, D_MODEL, D_MODEL, 1, 0, 0);
        gemm(Abf, Bbf, Dp[DBq] + i * BD, Qb, ROWS, D_MODEL, D_MODEL, 1.f, 0, 1, 0, 0, 0);
        cvtT(Dp[DWk] + i * WQ, Bbf, D_MODEL, D_MODEL, 1, 0, 0);
        gemm(Abf, Bbf, Dp[DBk] + i * BD, Kb, ROWS, D_MODEL, D_MODEL, 1.f, 0, 1, 0, 0, 0);
        cvtT(Dp[DWv] + i * WQ, Bbf, D_MODEL, D_MODEL, 1, 0, 0);
        gemm(Abf, Bbf, Dp[DBv] + i * BD, Vb, ROWS, D_MODEL, D_MODEL, 1.f, 0, 1, 0, 0, 0);
        cvt(Qb, Abf, ACT); cvt(Kb, Bbf, ACT);
        gemm(Abf, Bbf, nullptr, Sb, SEQ_LEN, SEQ_LEN, D_MODEL, AATT, 0, BATCH, sQK, sQK, sSS);
        softmax512<<<dim3(ROWS), 128, 0, stream>>>(Sb, Sb);
        cvt(Sb, Abf, SCO); cvtT(Vb, Bbf, SEQ_LEN, D_MODEL, BATCH, sQK, sQK);
        gemm(Abf, Bbf, nullptr, Qb, SEQ_LEN, D_MODEL, SEQ_LEN, 1.f, 0, BATCH, sSS, sQK, sQK);
        cvt(Qb, Abf, ACT); cvtT(Dp[DWu] + i * WQ, Bbf, D_MODEL, D_MODEL, 1, 0, 0);
        gemm(Abf, Bbf, Dp[DBu] + i * BD, Vb, ROWS, D_MODEL, D_MODEL, 1.f, 0, 1, 0, 0, 0);
        add_layernorm<<<dim3(ROWS), 256, 0, stream>>>(X, Vb, Dp[DLn1g] + i * BD, Dp[DLn1b] + i * BD, N1);
        // cross-attention vs last-encoder K/V
        cvt(N1, Abf, ACT); cvtT(Dp[DWedq] + i * WQ, Bbf, D_MODEL, D_MODEL, 1, 0, 0);
        gemm(Abf, Bbf, Dp[DBedq] + i * BD, Qb, ROWS, D_MODEL, D_MODEL, 1.f, 0, 1, 0, 0, 0);
        float* Sc = last ? OUT_S : Sb;
        cvt(Qb, Abf, ACT); cvt(EK, Bbf, ACT);
        gemm(Abf, Bbf, nullptr, Sc, SEQ_LEN, SEQ_LEN, D_MODEL, AATT, 0, BATCH, sQK, sQK, sSS);
        softmax512<<<dim3(ROWS), 128, 0, stream>>>(Sc, Sc);
        cvt(Sc, Abf, SCO); cvtT(EV, Bbf, SEQ_LEN, D_MODEL, BATCH, sQK, sQK);
        gemm(Abf, Bbf, nullptr, Qb, SEQ_LEN, D_MODEL, SEQ_LEN, 1.f, 0, BATCH, sSS, sQK, sQK);
        cvt(Qb, Abf, ACT); cvtT(Dp[DWedu] + i * WQ, Bbf, D_MODEL, D_MODEL, 1, 0, 0);
        gemm(Abf, Bbf, Dp[DBedu] + i * BD, Vb, ROWS, D_MODEL, D_MODEL, 1.f, 0, 1, 0, 0, 0);
        add_layernorm<<<dim3(ROWS), 256, 0, stream>>>(Vb, N1, Dp[DLn2g] + i * BD, Dp[DLn2b] + i * BD, N2);
        // FFN
        cvt(N2, Abf, ACT); cvtT(Dp[DW1] + i * W1S, Bbf, D_MODEL, FF_DIM, 1, 0, 0);
        gemm(Abf, Bbf, Dp[DB1] + i * BF, FFb, ROWS, FF_DIM, D_MODEL, 1.f, 1, 1, 0, 0, 0);
        cvt(FFb, Abf, FFA); cvtT(Dp[DW2] + i * W1S, Bbf, FF_DIM, D_MODEL, 1, 0, 0);
        gemm(Abf, Bbf, Dp[DB2] + i * BD, Qb, ROWS, D_MODEL, FF_DIM, 1.f, 0, 1, 0, 0, 0);
        float* Yo = last ? OUT_Y : X;
        add_layernorm<<<dim3(ROWS), 256, 0, stream>>>(N2, Qb, Dp[DLn3g] + i * BD, Dp[DLn3b] + i * BD, Yo);
    }
}
